// Attention_14568529068500
// MI455X (gfx1250) — matrix operand order repaired and checked by reading, not compiled, not run on hardware
//
#include <hip/hip_runtime.h>

// ---------------------------------------------------------------------------
// MI455X (gfx1250) bf16-WMMA attention pipeline:
//   qh = (q*Wq)/sqrt(dk), kh = k*Wk, vhT = (v*Wv)^T   (bf16)
//   flash attention (causal, online softmax, 32 q-rows/wave,
//                    unmasked main loop + single masked tail block,
//                    batched butterfly reductions, row-sums via P x ones WMMA)
//   out = O * w_o  (f32)
// ---------------------------------------------------------------------------

typedef __attribute__((ext_vector_type(16))) unsigned short v16u;
typedef __attribute__((ext_vector_type(16))) __bf16         v16bf;
typedef __attribute__((ext_vector_type(2)))  __bf16         v2bf;
typedef __attribute__((ext_vector_type(8)))  float          v8f;
typedef __attribute__((ext_vector_type(4)))  float          v4fl;
typedef __attribute__((ext_vector_type(2)))  unsigned int   v2ui;
typedef __attribute__((ext_vector_type(4)))  unsigned int   v4ui;

#define NB   4       // batch
#define NH   16      // heads
#define NS   2048    // sequence
#define DM   1024    // d_model
#define HD   64      // head dim
#define MROWS (NB * NS)          // 8192
#define SCALE 0.125f             // 1/sqrt(64)
#define NEGF  (-1e9f)

static __device__ __forceinline__ unsigned short f2bf(float f) {
  unsigned u = __builtin_bit_cast(unsigned, f);
  u = (u + 0x7FFFu + ((u >> 16) & 1u)) >> 16;     // round-to-nearest-even
  return (unsigned short)u;
}
// pack two floats to 2xbf16 in one dword (hardware packed cvt when available)
static __device__ __forceinline__ unsigned pack2bf(float lo, float hi) {
#if __has_builtin(__builtin_amdgcn_cvt_pk_bf16_f32)
  v2bf r = __builtin_amdgcn_cvt_pk_bf16_f32(lo, hi);
  return __builtin_bit_cast(unsigned, r);
#else
  return (unsigned)f2bf(lo) | ((unsigned)f2bf(hi) << 16);
#endif
}
static __device__ __forceinline__ v8f vzero8() {
  v8f z;
#pragma unroll
  for (int i = 0; i < 8; ++i) z[i] = 0.f;
  return z;
}

// A-fragment (16x32, row-major src, element stride ld).
// ISA layout: lane m=lane&15, half=lane>>4; elems 0..7 -> K=8*half+0..7,
// elems 8..15 -> K=16+8*half+0..7.
static __device__ __forceinline__ v16u load_a_frag(const unsigned short* src, int ld) {
  const int lane = threadIdx.x & 31;
  const int m = lane & 15, hi = lane >> 4;
  const unsigned short* p0 = src + m * ld + 8 * hi;
  const unsigned short* p1 = p0 + 16;
  v16u a;
#pragma unroll
  for (int i = 0; i < 8; ++i) { a[i] = p0[i]; a[8 + i] = p1[i]; }
  return a;
}

// B-fragment (32x16 = KxN). srcT holds B^T rows (row n = column n of B).
// ISA layout: lane n=lane&15, half=lane>>4; elems 0..7 -> K=8*half+0..7,
// elems 8..15 -> K=16+8*half+0..7.
static __device__ __forceinline__ v16u load_b_frag(const unsigned short* srcT, int ld) {
  const int lane = threadIdx.x & 31;
  const int n = lane & 15, hi = lane >> 4;
  const unsigned short* p = srcT + n * ld + 8 * hi;
  v16u b;
#pragma unroll
  for (int i = 0; i < 16; ++i) b[i] = (i < 8) ? p[i] : p[i + 8];
  return b;
}

static __device__ __forceinline__ v8f wmma_bf16(v16u a, v16u b, v8f c) {
  return __builtin_amdgcn_wmma_f32_16x16x32_bf16(
      false, __builtin_bit_cast(v16bf, a),
      false, __builtin_bit_cast(v16bf, b),
      (short)0, c, false, false);
}

// --------------------------- weight prep -----------------------------------
// w [H][D][64] f32 -> wT [h*64+c][d] bf16  (combined per-head GEMM B^T)
__global__ void k_pack_w_heads(const float* __restrict__ w, unsigned short* __restrict__ wT) {
  int idx = blockIdx.x * blockDim.x + threadIdx.x;   // over 1024*1024
  int nn = idx >> 10;                                // h*64 + c
  int d  = idx & 1023;
  int h = nn >> 6, c = nn & 63;
  wT[idx] = f2bf(w[((size_t)(h << 10) + d) * 64 + c]);
}
// w_o [1024][1024] f32 -> woT [n][k] bf16 (transpose)
__global__ void k_pack_wo(const float* __restrict__ w, unsigned short* __restrict__ wT) {
  int idx = blockIdx.x * blockDim.x + threadIdx.x;
  int nn = idx >> 10, kk = idx & 1023;
  wT[idx] = f2bf(w[(size_t)kk * 1024 + nn]);
}

// --------------------------- tiled GEMM ------------------------------------
// C[8192][1024] = oscale * A[8192][1024] * B[1024][1024], BT[n][k] given.
// Workgroup: 256 thr / 8 waves, tile 128x128; wave tile 32x64 (8 WMMA/chunk).
// MODE 0: store bf16 [B][H][N][64]   (qh/kh)
// MODE 1: store bf16 [B][H][64][N]   (vh transposed; packed b128 stores)
// MODE 2: store f32  [M][1024]       (final output)
template <bool AF32, int MODE>
__global__ __launch_bounds__(256) void k_gemm_bt(const void* __restrict__ Ain,
                                                 const unsigned short* __restrict__ BT,
                                                 void* __restrict__ Out,
                                                 float oscale) {
  __shared__ unsigned short As[128 * 40];   // 128 rows x 32 k (pad 40)
  __shared__ unsigned short Bs[128 * 40];   // 128 cols x 32 k (pad 40)
  const int row0 = blockIdx.x * 128;
  const int n0   = blockIdx.y * 128;
  const int tid  = threadIdx.x;
  const int wave = __builtin_amdgcn_readfirstlane(tid >> 5);   // scalar
  const int wr = wave >> 1, wc = wave & 1;  // 4x2 waves -> 32x64 per wave
  const int lane = tid & 31, nn = lane & 15, hi = lane >> 4;

  v8f acc[2][4];
#pragma unroll
  for (int i = 0; i < 2; ++i)
#pragma unroll
    for (int j = 0; j < 4; ++j) acc[i][j] = vzero8();

  for (int k0 = 0; k0 < 1024; k0 += 32) {
#pragma unroll
    for (int it = 0; it < 4; ++it) {          // stage A: 128x32, 4-wide
      int idx = tid + it * 256;
      int r = idx >> 3, kv = (idx & 7) * 4;
      v2ui pk;
      if (AF32) {
        v4fl f = *(const v4fl*)((const float*)Ain + (size_t)(row0 + r) * 1024 + k0 + kv);
        pk[0] = pack2bf(f[0], f[1]);
        pk[1] = pack2bf(f[2], f[3]);
      } else {
        pk = *(const v2ui*)((const unsigned short*)Ain + (size_t)(row0 + r) * 1024 + k0 + kv);
      }
      *(v2ui*)&As[r * 40 + kv] = pk;
    }
#pragma unroll
    for (int it = 0; it < 4; ++it) {          // stage B: 128x32, 4-wide
      int idx = tid + it * 256;
      int r = idx >> 3, kv = (idx & 7) * 4;
      *(v2ui*)&Bs[r * 40 + kv] = *(const v2ui*)(BT + (size_t)(n0 + r) * 1024 + k0 + kv);
    }
    __syncthreads();
    v16u a0 = load_a_frag(&As[(wr * 32) * 40], 40);
    v16u a1 = load_a_frag(&As[(wr * 32 + 16) * 40], 40);
#pragma unroll
    for (int jt = 0; jt < 4; ++jt) {
      v16u bf = load_b_frag(&Bs[(wc * 64 + 16 * jt) * 40], 40);
      acc[0][jt] = wmma_bf16(a0, bf, acc[0][jt]);
      acc[1][jt] = wmma_bf16(a1, bf, acc[1][jt]);
    }
    __syncthreads();
  }

#pragma unroll
  for (int i = 0; i < 2; ++i)
#pragma unroll
    for (int jt = 0; jt < 4; ++jt) {
      if (MODE == 1) {
        // lane's 8 elements are 8 consecutive seq rows at fixed dv column:
        // pack to one 16-byte store into vT[(b,h,c)][seq].
        int R = row0 + wr * 32 + 16 * i + 8 * hi;        // seq base (j = 0)
        int C = n0 + wc * 64 + 16 * jt + nn;
        int b = R >> 11, nr = R & 2047;
        int h = C >> 6,  c  = C & 63;
        v4ui pk;
#pragma unroll
        for (int q2 = 0; q2 < 4; ++q2)
          pk[q2] = pack2bf(oscale * acc[i][jt][2 * q2], oscale * acc[i][jt][2 * q2 + 1]);
        *(v4ui*)&((unsigned short*)Out)[((size_t)(b * NH + h) * HD + c) * NS + nr] = pk;
      } else {
#pragma unroll
        for (int j = 0; j < 8; ++j) {
          int R = row0 + wr * 32 + 16 * i + j + 8 * hi;  // global row
          int C = n0 + wc * 64 + 16 * jt + nn;           // global col
          float val = oscale * acc[i][jt][j];
          if (MODE == 2) {
            ((float*)Out)[(size_t)R * 1024 + C] = val;
          } else {
            int b = R >> 11, nr = R & 2047;
            int h = C >> 6,  c  = C & 63;
            ((unsigned short*)Out)[(((size_t)(b * NH + h) * NS + nr) << 6) + c] = f2bf(val);
          }
        }
      }
    }
}

// --------------------------- flash attention --------------------------------
// One wave per 32-row Q tile (2 row-fragments reuse each K/V fragment).
// s-blocks of 32; causal online softmax; l accumulated via P x ones WMMA.
// Q was pre-scaled by 1/sqrt(dk) in the projection; mask only in tail block.
__global__ __launch_bounds__(128) void k_attn(const unsigned short* __restrict__ qh,
                                              const unsigned short* __restrict__ kh,
                                              const unsigned short* __restrict__ vT,
                                              unsigned short* __restrict__ O) {
  __shared__ unsigned short p_all[4 * 32 * 40];          // wave-private P tiles
  const int tid = threadIdx.x;
  const int wave = __builtin_amdgcn_readfirstlane(tid >> 5);   // scalar
  unsigned short* p_lds = &p_all[wave * 32 * 40];
  const int lane = tid & 31, n = lane & 15, hi = lane >> 4;

  const int tile = blockIdx.x * 4 + wave;                // scalar, 0..4095
  const int qt = tile & 63;
  const int bh = tile >> 6;                              // b*16+h
  const int q0 = qt << 5;
  const int b = bh >> 4, h = bh & 15;

  const unsigned short* qbase = qh + ((size_t)bh * NS + q0) * HD;
  const unsigned short* kbase = kh + (size_t)bh * NS * HD;
  const unsigned short* vbase = vT + (size_t)bh * HD * NS;

  v16u a_q[2][2];
#pragma unroll
  for (int rt = 0; rt < 2; ++rt) {
    a_q[rt][0] = load_a_frag(qbase + (size_t)(16 * rt) * HD, HD);       // dk 0..31
    a_q[rt][1] = load_a_frag(qbase + (size_t)(16 * rt) * HD + 32, HD);  // dk 32..63
  }
  v16u ones;
#pragma unroll
  for (int i = 0; i < 16; ++i) ones[i] = 0x3F80;         // bf16(1.0)

  v8f o_acc[2][4], l_acc[2];
  float m_i[2][8];
#pragma unroll
  for (int rt = 0; rt < 2; ++rt) {
    l_acc[rt] = vzero8();
#pragma unroll
    for (int d = 0; d < 4; ++d) o_acc[rt][d] = vzero8();
#pragma unroll
    for (int j = 0; j < 8; ++j) m_i[rt][j] = -1e30f;
  }

  auto process_block = [&](int s0, bool need_mask) {
    // ---- S = Q K^T for cols [s0, s0+32): shared K fragments, 8 WMMAs
    v16u bk[2][2];
#pragma unroll
    for (int t = 0; t < 2; ++t) {
      const unsigned short* kb = kbase + (size_t)(s0 + 16 * t) * HD;
      bk[t][0] = load_b_frag(kb, HD);
      bk[t][1] = load_b_frag(kb + 32, HD);
    }
    v8f s_frag[2][2];
#pragma unroll
    for (int rt = 0; rt < 2; ++rt)
#pragma unroll
      for (int t = 0; t < 2; ++t) {
        v8f c = vzero8();
        c = wmma_bf16(a_q[rt][0], bk[t][0], c);
        c = wmma_bf16(a_q[rt][1], bk[t][1], c);
        s_frag[rt][t] = c;
      }

    // ---- row max: local max, then step-major batched butterflies
    float mx[2][8];
#pragma unroll
    for (int rt = 0; rt < 2; ++rt)
#pragma unroll
      for (int j = 0; j < 8; ++j) {
        float a0 = s_frag[rt][0][j];
        float a1 = s_frag[rt][1][j];
        if (need_mask) {
          const int row = q0 + 16 * rt + j + 8 * hi;
          if (s0 + n > row)      a0 = NEGF;
          if (s0 + 16 + n > row) a1 = NEGF;
          s_frag[rt][0][j] = a0;
          s_frag[rt][1][j] = a1;
        }
        mx[rt][j] = fmaxf(a0, a1);
      }
#pragma unroll
    for (int st = 1; st <= 8; st <<= 1) {          // 16 bpermutes per step
      float t[2][8];
#pragma unroll
      for (int rt = 0; rt < 2; ++rt)
#pragma unroll
        for (int j = 0; j < 8; ++j) t[rt][j] = __shfl_xor(mx[rt][j], st, 16);
#pragma unroll
      for (int rt = 0; rt < 2; ++rt)
#pragma unroll
        for (int j = 0; j < 8; ++j) mx[rt][j] = fmaxf(mx[rt][j], t[rt][j]);
    }

    // ---- exponentiate + rescale; row sums come from the ones-WMMA
#pragma unroll
    for (int rt = 0; rt < 2; ++rt) {
      float alpha[8];
#pragma unroll
      for (int j = 0; j < 8; ++j) {
        const float mnew = fmaxf(m_i[rt][j], mx[rt][j]);
        alpha[j] = __expf(m_i[rt][j] - mnew);
        m_i[rt][j] = mnew;
        const unsigned pk = pack2bf(__expf(s_frag[rt][0][j] - mnew),
                                    __expf(s_frag[rt][1][j] - mnew));
        p_lds[(16 * rt + j + 8 * hi) * 40 + n]      = (unsigned short)pk;
        p_lds[(16 * rt + j + 8 * hi) * 40 + 16 + n] = (unsigned short)(pk >> 16);
      }
#pragma unroll
      for (int j = 0; j < 8; ++j) l_acc[rt][j] *= alpha[j];
#pragma unroll
      for (int d = 0; d < 4; ++d)
#pragma unroll
        for (int j = 0; j < 8; ++j) o_acc[rt][d][j] *= alpha[j];
    }

    // ---- O += P*V, l += P*1  (shared V fragments; DS in-order per wave)
    v16u a_p[2];
    a_p[0] = load_a_frag(p_lds, 40);
    a_p[1] = load_a_frag(p_lds + 16 * 40, 40);
    l_acc[0] = wmma_bf16(a_p[0], ones, l_acc[0]);
    l_acc[1] = wmma_bf16(a_p[1], ones, l_acc[1]);
#pragma unroll
    for (int d = 0; d < 4; ++d) {
      v16u bv = load_b_frag(vbase + (size_t)(16 * d) * NS + s0, NS);
      o_acc[0][d] = wmma_bf16(a_p[0], bv, o_acc[0][d]);
      o_acc[1][d] = wmma_bf16(a_p[1], bv, o_acc[1][d]);
    }
  };

  // blocks strictly below the diagonal are never masked (s0+31 <= q0)
  for (int s0 = 0; s0 < q0; s0 += 32) process_block(s0, false);
  process_block(q0, true);                               // diagonal tail block

  // ---- normalize (v_rcp) + store O[b][row][h*64+col] (bf16)
#pragma unroll
  for (int rt = 0; rt < 2; ++rt) {
    float inv[8];
#pragma unroll
    for (int j = 0; j < 8; ++j) inv[j] = __builtin_amdgcn_rcpf(l_acc[rt][j]);
#pragma unroll
    for (int d = 0; d < 4; ++d)
#pragma unroll
      for (int j = 0; j < 8; ++j) {
        const int row = q0 + 16 * rt + j + 8 * hi;
        const int col = h * HD + 16 * d + n;
        O[((size_t)b * NS + row) * DM + col] = f2bf(o_acc[rt][d][j] * inv[j]);
      }
  }
}

// --------------------------- launcher ---------------------------------------
extern "C" void kernel_launch(void* const* d_in, const int* in_sizes, int n_in,
                              void* d_out, int out_size, void* d_ws, size_t ws_size,
                              hipStream_t stream) {
  (void)in_sizes; (void)n_in; (void)out_size; (void)ws_size;
  const float* q   = (const float*)d_in[0];
  const float* k   = (const float*)d_in[1];
  const float* v   = (const float*)d_in[2];
  const float* w_q = (const float*)d_in[3];
  const float* w_k = (const float*)d_in[4];
  const float* w_v = (const float*)d_in[5];
  const float* w_o = (const float*)d_in[6];
  float* out = (float*)d_out;

  char* ws = (char*)d_ws;
  const size_t WSZ = (size_t)1024 * 1024 * 2;                 // 2 MB per weight
  const size_t ASZ = (size_t)NB * NH * NS * HD * 2;           // 16 MB per act
  unsigned short* wqT = (unsigned short*)(ws);
  unsigned short* wkT = (unsigned short*)(ws + WSZ);
  unsigned short* wvT = (unsigned short*)(ws + 2 * WSZ);
  unsigned short* woT = (unsigned short*)(ws + 3 * WSZ);
  unsigned short* qhB = (unsigned short*)(ws + 4 * WSZ);
  unsigned short* khB = (unsigned short*)(ws + 4 * WSZ + ASZ);
  unsigned short* vtB = (unsigned short*)(ws + 4 * WSZ + 2 * ASZ);
  unsigned short* oB  = (unsigned short*)(ws + 4 * WSZ + 3 * ASZ);

  // 1) pack weights to bf16 transposed layouts
  k_pack_w_heads<<<4096, 256, 0, stream>>>(w_q, wqT);
  k_pack_w_heads<<<4096, 256, 0, stream>>>(w_k, wkT);
  k_pack_w_heads<<<4096, 256, 0, stream>>>(w_v, wvT);
  k_pack_wo<<<4096, 256, 0, stream>>>(w_o, woT);

  // 2) QKV projections (Q pre-scaled by 1/sqrt(dk))
  dim3 gg(MROWS / 128, 1024 / 128);
  k_gemm_bt<true, 0><<<gg, 256, 0, stream>>>((const void*)q, wqT, (void*)qhB, SCALE);
  k_gemm_bt<true, 0><<<gg, 256, 0, stream>>>((const void*)k, wkT, (void*)khB, 1.0f);
  k_gemm_bt<true, 1><<<gg, 256, 0, stream>>>((const void*)v, wvT, (void*)vtB, 1.0f);

  // 3) causal flash attention (one wave per 32-row Q tile)
  k_attn<<<(NB * NH * (NS / 32)) / 4, 128, 0, stream>>>(qhB, khB, vtB, oB);

  // 4) output projection (f32 result)
  k_gemm_bt<false, 2><<<gg, 256, 0, stream>>>((const void*)oB, woT, (void*)out, 1.0f);
}
